// SFA_67310727462952
// MI455X (gfx1250) — matrix operand order repaired and checked by reading, not compiled, not run on hardware
//
#include <hip/hip_runtime.h>
#include <hip/hip_bf16.h>

#if __has_include(<hip/amd_detail/amd_gfx1250_TDM.h>)
#define TDM_6ARG 1
#else
#define TDM_6ARG 0
#endif

// ---------------- types ----------------
typedef __attribute__((ext_vector_type(16))) _Float16 v16h;
typedef __attribute__((ext_vector_type(8)))  _Float16 v8h;
typedef __attribute__((ext_vector_type(8)))  float    v8f;
typedef __attribute__((ext_vector_type(4)))  unsigned int u32x4;
typedef __attribute__((ext_vector_type(8)))  int      i32x8;
typedef __attribute__((ext_vector_type(4)))  int      i32x4;

// ---------------- problem constants ----------------
static constexpr int Bn   = 2;      // batch
static constexpr int C0   = 128;    // input channels of x
static constexpr int Cc   = 64;     // dim
static constexpr int Hh   = 64, Wh = 64;
static constexpr int HWc  = Hh * Wh;       // 4096  (n)
static constexpr int Hm   = 32, Wm = 32;
static constexpr int HWm  = Hm * Wm;       // 1024  (m)
static constexpr int NHn  = 8;      // heads
static constexpr int HID  = 256;    // hidden
static constexpr int C2   = 512;    // 2*hid

static inline int cdiv(int a, int b) { return (a + b - 1) / b; }

__device__ __forceinline__ float gelu_f(float x) {
    return 0.5f * x * (1.0f + erff(x * 0.70710678118654752440f));
}

// ---------------- weight pack / convert ----------------

// pack OIHW f32 weights -> [tap][Cout][Cin] f16
__global__ void k_pack(const float* __restrict__ w, _Float16* __restrict__ o,
                       int Cout, int Cin, int KHW) {
    int idx = blockIdx.x * blockDim.x + threadIdx.x;
    int total = Cout * Cin * KHW;
    if (idx >= total) return;
    int ci = idx % Cin; int r = idx / Cin;
    int co = r % Cout;  int tap = r / Cout;
    o[idx] = (_Float16)w[(co * Cin + ci) * KHW + tap];
}

__global__ void k_cvt(const float* __restrict__ s, _Float16* __restrict__ d, int n) {
    int i = blockIdx.x * blockDim.x + threadIdx.x;
    if (i < n) d[i] = (_Float16)s[i];
}

// ---------------- WMMA 1x1 conv with TDM (async tensor load to LDS) ----------------
// X: [B,Cin,HW] f16, Wp: [Cout,Cin] f16.  Block = 4 waves, one shared B tile
// [Cin x 16] staged in LDS by the Tensor Data Mover (one tensor_load per block),
// each wave computes a different 16-row Cout tile.  No padding -> tile is a pure
// rectangular window of the [Cin, HW] tensor, exactly TDM's 2D case.
template<int CIN>
__global__ __launch_bounds__(128) void k_conv1x1_tdm(
    const _Float16* __restrict__ X, const _Float16* __restrict__ Wp,
    const float* __restrict__ bias, float* __restrict__ Out, int Cout) {
    __shared__ _Float16 tileB[CIN * 16];           // [k][n]
    const int tid  = threadIdx.x;
    const int lane = tid & 31;
    const int wrp  = tid >> 5;
    const int n0   = blockIdx.x * 16;
    const int m0   = (blockIdx.y * 4 + wrp) * 16;
    const int b    = blockIdx.z;
    const int nl   = lane & 15;
    const int half = lane >> 4;

    if (wrp == 0) {
        // Build D# (groups 0/1) per CDNA5 ISA 8.3/8.4.  2D tensor [CIN, HW],
        // tile [CIN, 16] starting at column n0, data_size = 2 bytes.
        unsigned long long ga =
            (unsigned long long)(const void*)(X + ((size_t)b * CIN) * HWc + n0);
        unsigned lds_off = (unsigned)(unsigned long long)(const void*)&tileB[0];
        u32x4 g0;
        g0[0] = 1u;                                   // count=1, is_restore=0
        g0[1] = lds_off;                              // lds_addr (bytes)
        g0[2] = (unsigned)(ga & 0xffffffffull);       // global_addr[31:0]
        g0[3] = (unsigned)((ga >> 32) & 0x01ffffffull) | 0x80000000u; // [56:32]|type=2
        i32x8 g1;
        g1[0] = (int)(1u << 16);                      // data_size=1 (2B), no flags
        g1[1] = (int)((unsigned)HWc << 16);           // tensor_dim0[15:0] @ bits63:48
        g1[2] = (int)((unsigned)CIN << 16);           // dim0[31:16]=0, tensor_dim1 lo16
        g1[3] = (int)(16u << 16);                     // tensor_dim1 hi=0, tile_dim0=16
        g1[4] = (int)(unsigned)CIN;                   // tile_dim1=CIN, tile_dim2=0
        g1[5] = (int)(unsigned)HWc;                   // tensor_dim0_stride[31:0]
        g1[6] = 0;                                    // stride hi, dim1_stride lo
        g1[7] = 0;
        i32x4 z4 = {};
#if TDM_6ARG
        i32x8 z8 = {};
        __builtin_amdgcn_tensor_load_to_lds(g0, g1, z4, z4, z8, 0);
#else
        __builtin_amdgcn_tensor_load_to_lds(g0, g1, z4, z4, 0);
#endif
        __builtin_amdgcn_s_wait_tensorcnt(0);
    }
    __syncthreads();

    const _Float16* Wt = Wp + (size_t)(m0 + nl) * CIN;
    v8f acc = {};
    #pragma unroll
    for (int k0 = 0; k0 < CIN; k0 += 32) {
        const _Float16* ar = Wt + k0 + (half << 3);
        v8h alo = *(const v8h*)(ar);
        v8h ahi = *(const v8h*)(ar + 16);
        v16h a = __builtin_shufflevector(alo, ahi,
                 0,1,2,3,4,5,6,7,8,9,10,11,12,13,14,15);
        v16h bv;
        #pragma unroll
        for (int j = 0; j < 16; ++j)
            bv[j] = tileB[(k0 + (half << 3) + (j & 7) + ((j >> 3) << 4)) * 16 + nl];
        acc = __builtin_amdgcn_wmma_f32_16x16x32_f16(
                false, a, false, bv, (short)0, acc, false, false);
    }
    float* Outb = Out + (size_t)b * Cout * HWc;
    #pragma unroll
    for (int r = 0; r < 8; ++r) {
        int m = m0 + r + (half << 3);
        float v = acc[r];
        if (bias) v += bias[m];
        Outb[(size_t)m * HWc + n0 + nl] = v;
    }
}

// ---------------- WMMA KxK dense conv (Cin=Cout=64), LDS-staged B ----------------
// Activation tile staged transposed [n][k] (row stride 72 halfs: 16B-aligned,
// bank-spread) so each lane's B fragment is two ds_load_b128.  Per-tap iy check
// is uniform per block (16-aligned tiles inside a 64-wide row) -> uniform skip.
__global__ __launch_bounds__(128) void k_convKxK_wmma(
    const _Float16* __restrict__ X, const _Float16* __restrict__ Wp,
    const float* __restrict__ bias, float* __restrict__ Out,
    int KH, int KW, int pad, int dil) {
    constexpr int CIN = 64, COUT = 64, LDW = 72;
    __shared__ _Float16 tbT[16 * LDW];
    const int tid  = threadIdx.x;
    const int lane = tid & 31;
    const int wrp  = tid >> 5;
    const int n0   = blockIdx.x * 16;
    const int b    = blockIdx.z;
    const int m0   = wrp * 16;
    const int nl   = lane & 15;
    const int half = lane >> 4;
    const int oy   = n0 / Wh, ox0 = n0 % Wh;       // uniform per block
    const _Float16* Xb = X + ((size_t)b * CIN) * HWc;
    const int lr = tid >> 2;            // 0..31: k row (two passes)
    const int li = (tid & 3) * 4;       // starting col in tile
    v8f acc = {};
    for (int kh = 0; kh < KH; ++kh) {
        const int iy = oy + kh * dil - pad;
        if (iy < 0 || iy >= Hh) continue;          // uniform whole-tap skip
        for (int kw = 0; kw < KW; ++kw) {
            const int dxc = ox0 + kw * dil - pad;
            const int tap = kh * KW + kw;
            __syncthreads();
            #pragma unroll
            for (int rr = 0; rr < CIN; rr += 32) {
                const int k = rr + lr;
                const _Float16* xr = Xb + (size_t)k * HWc + iy * Wh;
                #pragma unroll
                for (int ii = 0; ii < 4; ++ii) {
                    const int i  = li + ii;
                    const int ix = dxc + i;
                    _Float16 v = (_Float16)0.0f;
                    if (ix >= 0 && ix < Wh) v = xr[ix];
                    tbT[i * LDW + k] = v;
                }
            }
            __syncthreads();
            const _Float16* Wt = Wp + ((size_t)tap * COUT + (m0 + nl)) * CIN;
            #pragma unroll
            for (int k0 = 0; k0 < CIN; k0 += 32) {
                const _Float16* ar = Wt + k0 + (half << 3);
                v8h alo = *(const v8h*)(ar);
                v8h ahi = *(const v8h*)(ar + 16);
                v16h a = __builtin_shufflevector(alo, ahi,
                         0,1,2,3,4,5,6,7,8,9,10,11,12,13,14,15);
                const _Float16* br = &tbT[nl * LDW + k0 + (half << 3)];
                v8h blo = *(const v8h*)(br);
                v8h bhi = *(const v8h*)(br + 16);
                v16h bv = __builtin_shufflevector(blo, bhi,
                          0,1,2,3,4,5,6,7,8,9,10,11,12,13,14,15);
                acc = __builtin_amdgcn_wmma_f32_16x16x32_f16(
                        false, a, false, bv, (short)0, acc, false, false);
            }
        }
    }
    float* Outb = Out + (size_t)b * COUT * HWc;
    #pragma unroll
    for (int r = 0; r < 8; ++r) {
        int m = m0 + r + (half << 3);
        float v = acc[r];
        if (bias) v += bias[m];
        Outb[(size_t)m * HWc + n0 + nl] = v;
    }
}

// ---------------- supporting kernels ----------------

// depthwise 3x3, pad=1, stride 1 or 2, no bias
__global__ void k_dw3x3(const float* __restrict__ X, const float* __restrict__ Wd,
                        float* __restrict__ Out, int Bc, int Cch, int H, int W, int stride) {
    int Ho = H / stride, Wo = W / stride;
    int total = Bc * Cch * Ho * Wo;
    int idx = blockIdx.x * blockDim.x + threadIdx.x;
    if (idx >= total) return;
    int ox = idx % Wo; int t = idx / Wo;
    int oy = t % Ho;   t /= Ho;
    int c  = t % Cch;  int b = t / Cch;
    const float* Xb = X + ((size_t)b * Cch + c) * H * W;
    const float* w  = Wd + c * 9;
    float acc = 0.f;
    #pragma unroll
    for (int kh = 0; kh < 3; ++kh) {
        int iy = oy * stride + kh - 1;
        if (iy < 0 || iy >= H) continue;
        #pragma unroll
        for (int kw = 0; kw < 3; ++kw) {
            int ix = ox * stride + kw - 1;
            if (ix < 0 || ix >= W) continue;
            acc += Xb[iy * W + ix] * w[kh * 3 + kw];
        }
    }
    Out[((size_t)b * Cch + c) * Ho * Wo + oy * Wo + ox] = acc;
}

// BatchNorm (training stats, biased var) + optional activation (0=none,1=gelu,2=relu)
__global__ __launch_bounds__(256) void k_bn(const float* __restrict__ X, float* __restrict__ Out,
        const float* __restrict__ g, const float* __restrict__ bta,
        int Bc, int Cch, int HWt, int act) {
    __shared__ float s1[256], s2[256];
    int c = blockIdx.x, tid = threadIdx.x;
    int cnt = Bc * HWt;
    float s = 0.f, ss = 0.f;
    for (int i = tid; i < cnt; i += 256) {
        int b = i / HWt, p = i % HWt;
        float v = X[((size_t)b * Cch + c) * HWt + p];
        s += v; ss += v * v;
    }
    s1[tid] = s; s2[tid] = ss; __syncthreads();
    for (int o = 128; o > 0; o >>= 1) {
        if (tid < o) { s1[tid] += s1[tid + o]; s2[tid] += s2[tid + o]; }
        __syncthreads();
    }
    float mean = s1[0] / (float)cnt;
    float var  = s2[0] / (float)cnt - mean * mean;
    float inv  = rsqrtf(var + 1e-5f) * g[c];
    float bb   = bta[c];
    for (int i = tid; i < cnt; i += 256) {
        int b = i / HWt, p = i % HWt;
        size_t o2 = ((size_t)b * Cch + c) * HWt + p;
        float v = (X[o2] - mean) * inv + bb;
        if (act == 1) v = gelu_f(v);
        else if (act == 2) v = fmaxf(v, 0.f);
        Out[o2] = v;
    }
}

// per-pixel LayerNorm over 64 channels
__global__ void k_ln(const float* __restrict__ X, float* __restrict__ Out,
                     const float* __restrict__ w, const float* __restrict__ b2,
                     int Bc, int HWt) {
    int idx = blockIdx.x * blockDim.x + threadIdx.x;
    int total = Bc * HWt;
    if (idx >= total) return;
    int b = idx / HWt, p = idx % HWt;
    const float* xb = X + (size_t)b * Cc * HWt + p;
    float* ob       = Out + (size_t)b * Cc * HWt + p;
    float mu = 0.f;
    for (int c = 0; c < Cc; ++c) mu += xb[(size_t)c * HWt];
    mu *= (1.f / Cc);
    float var = 0.f;
    for (int c = 0; c < Cc; ++c) { float d = xb[(size_t)c * HWt] - mu; var += d * d; }
    var *= (1.f / Cc);
    float inv = rsqrtf(var + 1e-5f);
    for (int c = 0; c < Cc; ++c)
        ob[(size_t)c * HWt] = (xb[(size_t)c * HWt] - mu) * inv * w[c] + b2[c];
}

// in-place L2 normalization of rows of length L
__global__ __launch_bounds__(256) void k_l2(float* __restrict__ X, int L) {
    __shared__ float sm[256];
    int row = blockIdx.x, tid = threadIdx.x;
    float* base = X + (size_t)row * L;
    float ss = 0.f;
    for (int i = tid; i < L; i += 256) { float v = base[i]; ss += v * v; }
    sm[tid] = ss; __syncthreads();
    for (int o = 128; o > 0; o >>= 1) {
        if (tid < o) sm[tid] += sm[tid + o];
        __syncthreads();
    }
    float sc = 1.f / fmaxf(sqrtf(sm[0]), 1e-12f);
    for (int i = tid; i < L; i += 256) base[i] *= sc;
}

// R[bh][c][d] = sum_p X[b, h*8+c, p] * Y[b, h*8+d, p]
__global__ __launch_bounds__(256) void k_rowdot(const float* __restrict__ X,
        const float* __restrict__ Y, float* __restrict__ R, int L) {
    __shared__ float sm[256];
    int bh = blockIdx.x;
    int b = bh / NHn, h = bh % NHn;
    int t = threadIdx.x;
    int pair = t >> 2, sub = t & 3;
    int c = pair >> 3, d = pair & 7;
    const float* xr = X + ((size_t)b * Cc + h * 8 + c) * L;
    const float* yr = Y + ((size_t)b * Cc + h * 8 + d) * L;
    float acc = 0.f;
    for (int p = sub; p < L; p += 4) acc += xr[p] * yr[p];
    sm[t] = acc; __syncthreads();
    if (sub == 0) R[bh * 64 + pair] = sm[t] + sm[t + 1] + sm[t + 2] + sm[t + 3];
}

// S = (softmax(Gx*t1) @ softmax(Gy*t2) @ P) * t3   (all 8x8 per (b,h))
__global__ __launch_bounds__(64) void k_attn(const float* __restrict__ Gx,
        const float* __restrict__ Gy, const float* __restrict__ P,
        const float* __restrict__ t1, const float* __restrict__ t2,
        const float* __restrict__ t3, float* __restrict__ S) {
    __shared__ float ax[8][8], ay[8][8], A[8][8], Pl[8][8];
    int bh = blockIdx.x;
    int h = bh % NHn;
    int t = threadIdx.x, r = t >> 3, cl = t & 7;
    ax[r][cl] = Gx[bh * 64 + t] * t1[h];
    ay[r][cl] = Gy[bh * 64 + t] * t2[h];
    Pl[r][cl] = P[bh * 64 + t];
    __syncthreads();
    if (t < 16) {
        float (*mrow)[8] = (t < 8) ? ax : ay;
        int rr = t & 7;
        float mx = mrow[rr][0];
        for (int d = 1; d < 8; ++d) mx = fmaxf(mx, mrow[rr][d]);
        float ev[8], sum = 0.f;
        for (int d = 0; d < 8; ++d) { ev[d] = expf(mrow[rr][d] - mx); sum += ev[d]; }
        float inv = 1.f / sum;
        for (int d = 0; d < 8; ++d) mrow[rr][d] = ev[d] * inv;
    }
    __syncthreads();
    float a = 0.f;
    for (int k = 0; k < 8; ++k) a += ax[r][k] * ay[k][cl];
    A[r][cl] = a;
    __syncthreads();
    float s2 = 0.f;
    for (int k = 0; k < 8; ++k) s2 += A[r][k] * Pl[k][cl];
    S[bh * 64 + t] = s2 * t3[h];
}

// Out[b, h*8+c, p] = sum_e S[bh][c][e] * VY[b, h*8+e, p]
__global__ __launch_bounds__(256) void k_applyS(const float* __restrict__ S,
        const float* __restrict__ VY, float* __restrict__ Out, int L) {
    __shared__ float sl[64];
    int tid = threadIdx.x;
    int h = blockIdx.y, b = blockIdx.z;
    if (tid < 64) sl[tid] = S[((size_t)b * NHn + h) * 64 + tid];
    __syncthreads();
    int p = blockIdx.x * 256 + tid;
    if (p >= L) return;
    float vv[8];
    #pragma unroll
    for (int e = 0; e < 8; ++e) vv[e] = VY[((size_t)b * Cc + h * 8 + e) * L + p];
    #pragma unroll
    for (int c = 0; c < 8; ++c) {
        float acc = 0.f;
        #pragma unroll
        for (int e = 0; e < 8; ++e) acc += sl[c * 8 + e] * vv[e];
        Out[((size_t)b * Cc + h * 8 + c) * L + p] = acc;
    }
}

__global__ void k_add(const float* __restrict__ a, const float* __restrict__ b,
                      float* __restrict__ o, int n) {
    int i = blockIdx.x * blockDim.x + threadIdx.x;
    if (i < n) o[i] = a[i] + b[i];
}

__global__ void k_fusepre(const float* __restrict__ a, const float* __restrict__ o_,
                          float* __restrict__ f, int n) {
    int i = blockIdx.x * blockDim.x + threadIdx.x;
    if (i < n) f[i] = a[i] + a[i] * o_[i];
}

// G[b,c,p] = gelu(Z[b,c,p]) * Z[b,c+256,p]
__global__ void k_geglu(const float* __restrict__ Z, float* __restrict__ G, int HWt) {
    int idx = blockIdx.x * blockDim.x + threadIdx.x;
    int total = Bn * HID * HWt;
    if (idx >= total) return;
    int p = idx % HWt; int t = idx / HWt;
    int c = t % HID;   int b = t / HID;
    float z1 = Z[((size_t)b * C2 + c) * HWt + p];
    float z2 = Z[((size_t)b * C2 + HID + c) * HWt + p];
    G[((size_t)b * HID + c) * HWt + p] = gelu_f(z1) * z2;
}

// ---------------- host orchestration ----------------

enum {
    IN_X = 0, IN_Y = 1,
    IN_WQX0 = 2, IN_WQX1, IN_WQX2, IN_WQY0, IN_WQY1, IN_WQY2,
    IN_DWX1 = 8, IN_DWX2, IN_DWX3, IN_DWY1, IN_DWY2, IN_DWY3,
    IN_T1 = 14, IN_T2, IN_T3,
    IN_PROJ = 17, IN_FFN_IN = 18, IN_FFN_DW = 19, IN_FFN_OUT = 20,
    IN_LN_W = 21, IN_LN_B, IN_LNX_W, IN_LNX_B, IN_LNY_W, IN_LNY_B,
    IN_FUSE_W1 = 27, IN_FUSE_B1, IN_FUSE_W2, IN_FUSE_B2, IN_FUSE_G, IN_FUSE_BETA,
    IN_B0_W = 33, IN_B0_G1, IN_B0_B1,
    IN_B1_W1 = 36, IN_B1_W2, IN_B1_W3,
    IN_B2_W1 = 39, IN_B2_W2, IN_B2_W3, IN_B2_W4,
    IN_B1_G1 = 43, IN_B1_B1, IN_B1_G2, IN_B1_B2, IN_B1_G3, IN_B1_B3,
    IN_B2_G1 = 49, IN_B2_B1, IN_B2_G2, IN_B2_B2, IN_B2_G3, IN_B2_B3, IN_B2_G4, IN_B2_B4
};

static void conv1x1(hipStream_t st, const _Float16* X, const _Float16* Wp,
                    const float* bias, float* Out, int Cin, int Cout) {
    dim3 g(HWc / 16, Cout / 64, Bn);
    if (Cin == 64)       k_conv1x1_tdm<64 ><<<g, 128, 0, st>>>(X, Wp, bias, Out, Cout);
    else if (Cin == 128) k_conv1x1_tdm<128><<<g, 128, 0, st>>>(X, Wp, bias, Out, Cout);
    else                 k_conv1x1_tdm<256><<<g, 128, 0, st>>>(X, Wp, bias, Out, Cout);
}

static void convK(hipStream_t st, const _Float16* X, const _Float16* Wp,
                  const float* bias, float* Out, int KH, int KW, int pad, int dil) {
    dim3 g(HWc / 16, 1, Bn);
    k_convKxK_wmma<<<g, 128, 0, st>>>(X, Wp, bias, Out, KH, KW, pad, dil);
}

struct CSP {
    const _Float16 *wqh[6];
    const float *dw[6];
    const float *t1, *t2, *t3;
    const _Float16 *proj_h, *ffn_in_h, *ffn_out_h, *fuse_w1_h, *fuse_w2_h;
    const float *ffn_dw;
    const float *ln_w, *ln_b, *lnx_w, *lnx_b, *lny_w, *lny_b;
    const float *fuse_b1, *fuse_b2, *fuse_g, *fuse_beta;
    float *XN, *YN, *TA, *QX, *KX, *VX, *QY, *KY, *VY, *ATT, *OUT, *Z, *Z2, *G, *FF, *FB;
    float *GX, *GY, *PP, *SS;
    _Float16 *AHy, *AH256;
};

static void cross_strengthen(hipStream_t st, const CSP& P, const float* x,
                             const float* y, float* dst) {
    const int nTot = Bn * Cc * HWc;
    const int E = 256;
    k_ln<<<cdiv(Bn * HWc, E), E, 0, st>>>(x, P.XN, P.lnx_w, P.lnx_b, Bn, HWc);
    k_ln<<<cdiv(Bn * HWc, E), E, 0, st>>>(y, P.YN, P.lny_w, P.lny_b, Bn, HWc);
    // x-side q/k/v
    k_cvt<<<cdiv(nTot, E), E, 0, st>>>(P.XN, P.AHy, nTot);
    conv1x1(st, P.AHy, P.wqh[0], nullptr, P.TA, Cc, Cc);
    k_dw3x3<<<cdiv(Bn * Cc * HWc, E), E, 0, st>>>(P.TA, P.dw[0], P.QX, Bn, Cc, Hh, Wh, 1);
    conv1x1(st, P.AHy, P.wqh[1], nullptr, P.TA, Cc, Cc);
    k_dw3x3<<<cdiv(Bn * Cc * HWm, E), E, 0, st>>>(P.TA, P.dw[1], P.KX, Bn, Cc, Hh, Wh, 2);
    conv1x1(st, P.AHy, P.wqh[2], nullptr, P.TA, Cc, Cc);
    k_dw3x3<<<cdiv(Bn * Cc * HWm, E), E, 0, st>>>(P.TA, P.dw[2], P.VX, Bn, Cc, Hh, Wh, 2);
    // y-side
    k_cvt<<<cdiv(nTot, E), E, 0, st>>>(P.YN, P.AHy, nTot);
    conv1x1(st, P.AHy, P.wqh[3], nullptr, P.TA, Cc, Cc);
    k_dw3x3<<<cdiv(Bn * Cc * HWm, E), E, 0, st>>>(P.TA, P.dw[3], P.QY, Bn, Cc, Hh, Wh, 2);
    conv1x1(st, P.AHy, P.wqh[4], nullptr, P.TA, Cc, Cc);
    k_dw3x3<<<cdiv(Bn * Cc * HWc, E), E, 0, st>>>(P.TA, P.dw[4], P.KY, Bn, Cc, Hh, Wh, 1);
    conv1x1(st, P.AHy, P.wqh[5], nullptr, P.TA, Cc, Cc);
    k_dw3x3<<<cdiv(Bn * Cc * HWc, E), E, 0, st>>>(P.TA, P.dw[5], P.VY, Bn, Cc, Hh, Wh, 1);
    // l2 normalize q/k rows
    k_l2<<<Bn * Cc, E, 0, st>>>(P.QX, HWc);
    k_l2<<<Bn * Cc, E, 0, st>>>(P.KX, HWm);
    k_l2<<<Bn * Cc, E, 0, st>>>(P.QY, HWm);
    k_l2<<<Bn * Cc, E, 0, st>>>(P.KY, HWc);
    // attention, chain collapsed via associativity
    k_rowdot<<<Bn * NHn, E, 0, st>>>(P.QX, P.KY, P.GX, HWc);
    k_rowdot<<<Bn * NHn, E, 0, st>>>(P.QY, P.KX, P.GY, HWm);
    k_rowdot<<<Bn * NHn, E, 0, st>>>(P.VX, P.VX, P.PP, HWm);
    k_attn<<<Bn * NHn, 64, 0, st>>>(P.GX, P.GY, P.PP, P.t1, P.t2, P.t3, P.SS);
    k_applyS<<<dim3(HWc / 256, NHn, Bn), 256, 0, st>>>(P.SS, P.VY, P.ATT, HWc);
    // proj_out + residual
    k_cvt<<<cdiv(nTot, E), E, 0, st>>>(P.ATT, P.AHy, nTot);
    conv1x1(st, P.AHy, P.proj_h, nullptr, P.TA, Cc, Cc);
    k_add<<<cdiv(nTot, E), E, 0, st>>>(x, P.TA, P.OUT, nTot);
    // GEGLU FFN
    k_ln<<<cdiv(Bn * HWc, E), E, 0, st>>>(P.OUT, P.XN, P.ln_w, P.ln_b, Bn, HWc);
    k_cvt<<<cdiv(nTot, E), E, 0, st>>>(P.XN, P.AHy, nTot);
    conv1x1(st, P.AHy, P.ffn_in_h, nullptr, P.Z, Cc, C2);
    k_dw3x3<<<cdiv(Bn * C2 * HWc, E), E, 0, st>>>(P.Z, P.ffn_dw, P.Z2, Bn, C2, Hh, Wh, 1);
    k_geglu<<<cdiv(Bn * HID * HWc, E), E, 0, st>>>(P.Z2, P.G, HWc);
    k_cvt<<<cdiv(Bn * HID * HWc, E), E, 0, st>>>(P.G, P.AH256, Bn * HID * HWc);
    conv1x1(st, P.AH256, P.ffn_out_h, nullptr, P.TA, HID, Cc);
    k_add<<<cdiv(nTot, E), E, 0, st>>>(P.OUT, P.TA, P.OUT, nTot);
    // fuse: (x + x*out) -> 1x1(+b) -> 3x3(+b) -> BN -> ReLU
    k_fusepre<<<cdiv(nTot, E), E, 0, st>>>(x, P.OUT, P.FF, nTot);
    k_cvt<<<cdiv(nTot, E), E, 0, st>>>(P.FF, P.AHy, nTot);
    conv1x1(st, P.AHy, P.fuse_w1_h, P.fuse_b1, P.TA, Cc, Cc);
    k_cvt<<<cdiv(nTot, E), E, 0, st>>>(P.TA, P.AHy, nTot);
    convK(st, P.AHy, P.fuse_w2_h, P.fuse_b2, P.FB, 3, 3, 1, 1);
    k_bn<<<Cc, 256, 0, st>>>(P.FB, dst, P.fuse_g, P.fuse_beta, Bn, Cc, HWc, 2);
}

extern "C" void kernel_launch(void* const* d_in, const int* in_sizes, int n_in,
                              void* d_out, int out_size, void* d_ws, size_t ws_size,
                              hipStream_t stream) {
    hipStream_t st = stream;
    const int E = 256;
    auto F = [&](int i) { return (const float*)d_in[i]; };

    char* base = (char*)d_ws;
    size_t off = 0;
    auto alloc = [&](size_t bytes) -> void* {
        void* p = base + off;
        off = (off + bytes + 255) & ~(size_t)255;
        return p;
    };
    auto af = [&](size_t elems) { return (float*)alloc(elems * 4); };
    auto ah = [&](size_t elems) { return (_Float16*)alloc(elems * 2); };

    _Float16* wqh[6]; for (int i = 0; i < 6; ++i) wqh[i] = ah(Cc * Cc);
    _Float16* proj_h    = ah(Cc * Cc);
    _Float16* fuse_w1_h = ah(Cc * Cc);
    _Float16* fuse_w2_h = ah(Cc * Cc * 9);
    _Float16* ffn_in_h  = ah(C2 * Cc);
    _Float16* ffn_out_h = ah(Cc * HID);
    _Float16* b0_wh     = ah(Cc * C0);
    _Float16* b1_w1h    = ah(Cc * C0);
    _Float16* b2_w1h    = ah(Cc * C0);
    _Float16* b1_w2h    = ah(Cc * Cc * 49);
    _Float16* b2_w2h    = ah(Cc * Cc * 49);
    _Float16* b2_w3h    = ah(Cc * Cc * 49);
    _Float16* b1_w3h    = ah(Cc * Cc * 9);
    _Float16* b2_w4h    = ah(Cc * Cc * 9);
    _Float16* AHx   = ah((size_t)Bn * C0 * HWc);
    _Float16* AHy   = ah((size_t)Bn * Cc * HWc);
    _Float16* AH256 = ah((size_t)Bn * HID * HWc);
    const size_t nTot = (size_t)Bn * Cc * HWc;
    float *X0 = af(nTot), *X1 = af(nTot), *X2 = af(nTot);
    float *TA = af((size_t)Bn * C2 * HWc);
    float *TB = af(nTot);
    float *R1 = af(nTot), *R2 = af(nTot);
    float *XN = af(nTot), *YN = af(nTot);
    float *QX = af(nTot), *KY = af(nTot), *VY = af(nTot), *ATT = af(nTot);
    float *KX = af((size_t)Bn * Cc * HWm), *VX = af((size_t)Bn * Cc * HWm), *QY = af((size_t)Bn * Cc * HWm);
    float *OUT = af(nTot), *FF = af(nTot), *FB = af(nTot);
    float *Z  = af((size_t)Bn * C2 * HWc);
    float *Z2 = af((size_t)Bn * C2 * HWc);
    float *G  = af((size_t)Bn * HID * HWc);
    float *GX = af(Bn * NHn * 64), *GY = af(Bn * NHn * 64);
    float *PP = af(Bn * NHn * 64), *SS = af(Bn * NHn * 64);
    (void)ws_size; (void)n_in; (void)in_sizes; (void)out_size;

    auto pack = [&](const float* w, _Float16* o, int Co, int Ci, int KHW) {
        k_pack<<<cdiv(Co * Ci * KHW, E), E, 0, st>>>(w, o, Co, Ci, KHW);
    };
    for (int i = 0; i < 6; ++i) pack(F(IN_WQX0 + i), wqh[i], Cc, Cc, 1);
    pack(F(IN_PROJ),    proj_h,    Cc, Cc, 1);
    pack(F(IN_FUSE_W1), fuse_w1_h, Cc, Cc, 1);
    pack(F(IN_FUSE_W2), fuse_w2_h, Cc, Cc, 9);
    pack(F(IN_FFN_IN),  ffn_in_h,  C2, Cc, 1);
    pack(F(IN_FFN_OUT), ffn_out_h, Cc, HID, 1);
    pack(F(IN_B0_W),    b0_wh,     Cc, C0, 1);
    pack(F(IN_B1_W1),   b1_w1h,    Cc, C0, 1);
    pack(F(IN_B2_W1),   b2_w1h,    Cc, C0, 1);
    pack(F(IN_B1_W2),   b1_w2h,    Cc, Cc, 49);
    pack(F(IN_B2_W2),   b2_w2h,    Cc, Cc, 49);
    pack(F(IN_B2_W3),   b2_w3h,    Cc, Cc, 49);
    pack(F(IN_B1_W3),   b1_w3h,    Cc, Cc, 9);
    pack(F(IN_B2_W4),   b2_w4h,    Cc, Cc, 9);

    // ---- branches ----
    const int nX = Bn * C0 * HWc;
    k_cvt<<<cdiv(nX, E), E, 0, st>>>(F(IN_X), AHx, nX);
    conv1x1(st, AHx, b0_wh, nullptr, TA, C0, Cc);
    k_bn<<<Cc, 256, 0, st>>>(TA, X0, F(IN_B0_G1), F(IN_B0_B1), Bn, Cc, HWc, 0);
    // branch1
    conv1x1(st, AHx, b1_w1h, nullptr, TA, C0, Cc);
    k_bn<<<Cc, 256, 0, st>>>(TA, TB, F(IN_B1_G1), F(IN_B1_B1), Bn, Cc, HWc, 1);
    k_cvt<<<cdiv((int)nTot, E), E, 0, st>>>(TB, AHy, (int)nTot);
    convK(st, AHy, b1_w2h, nullptr, TA, 7, 7, 3, 1);
    k_bn<<<Cc, 256, 0, st>>>(TA, TB, F(IN_B1_G2), F(IN_B1_B2), Bn, Cc, HWc, 1);
    k_cvt<<<cdiv((int)nTot, E), E, 0, st>>>(TB, AHy, (int)nTot);
    convK(st, AHy, b1_w3h, nullptr, TA, 3, 3, 7, 7);
    k_bn<<<Cc, 256, 0, st>>>(TA, X1, F(IN_B1_G3), F(IN_B1_B3), Bn, Cc, HWc, 0);
    // branch2
    conv1x1(st, AHx, b2_w1h, nullptr, TA, C0, Cc);
    k_bn<<<Cc, 256, 0, st>>>(TA, TB, F(IN_B2_G1), F(IN_B2_B1), Bn, Cc, HWc, 1);
    k_cvt<<<cdiv((int)nTot, E), E, 0, st>>>(TB, AHy, (int)nTot);
    convK(st, AHy, b2_w2h, nullptr, TA, 7, 7, 3, 1);
    k_bn<<<Cc, 256, 0, st>>>(TA, TB, F(IN_B2_G2), F(IN_B2_B2), Bn, Cc, HWc, 1);
    k_cvt<<<cdiv((int)nTot, E), E, 0, st>>>(TB, AHy, (int)nTot);
    convK(st, AHy, b2_w3h, nullptr, TA, 7, 7, 3, 1);
    k_bn<<<Cc, 256, 0, st>>>(TA, TB, F(IN_B2_G3), F(IN_B2_B3), Bn, Cc, HWc, 1);
    k_cvt<<<cdiv((int)nTot, E), E, 0, st>>>(TB, AHy, (int)nTot);
    convK(st, AHy, b2_w4h, nullptr, TA, 3, 3, 7, 7);
    k_bn<<<Cc, 256, 0, st>>>(TA, X2, F(IN_B2_G4), F(IN_B2_B4), Bn, Cc, HWc, 0);

    // ---- three CrossStrengthen passes ----
    CSP P;
    for (int i = 0; i < 6; ++i) { P.wqh[i] = wqh[i]; P.dw[i] = F(IN_DWX1 + i); }
    P.t1 = F(IN_T1); P.t2 = F(IN_T2); P.t3 = F(IN_T3);
    P.proj_h = proj_h; P.ffn_in_h = ffn_in_h; P.ffn_out_h = ffn_out_h;
    P.fuse_w1_h = fuse_w1_h; P.fuse_w2_h = fuse_w2_h;
    P.ffn_dw = F(IN_FFN_DW);
    P.ln_w = F(IN_LN_W);  P.ln_b = F(IN_LN_B);
    P.lnx_w = F(IN_LNX_W); P.lnx_b = F(IN_LNX_B);
    P.lny_w = F(IN_LNY_W); P.lny_b = F(IN_LNY_B);
    P.fuse_b1 = F(IN_FUSE_B1); P.fuse_b2 = F(IN_FUSE_B2);
    P.fuse_g = F(IN_FUSE_G);   P.fuse_beta = F(IN_FUSE_BETA);
    P.XN = XN; P.YN = YN; P.TA = TA; P.QX = QX; P.KX = KX; P.VX = VX;
    P.QY = QY; P.KY = KY; P.VY = VY; P.ATT = ATT; P.OUT = OUT;
    P.Z = Z; P.Z2 = Z2; P.G = G; P.FF = FF; P.FB = FB;
    P.GX = GX; P.GY = GY; P.PP = PP; P.SS = SS;
    P.AHy = AHy; P.AH256 = AH256;

    cross_strengthen(st, P, X0, F(IN_Y), R1);
    cross_strengthen(st, P, R1, X1, R2);
    cross_strengthen(st, P, R2, X2, (float*)d_out);
}
